// GNN_137438954176
// MI455X (gfx1250) — compile-verified, matrix operand order checked by reading, not run on hardware
//
#include <hip/hip_runtime.h>

typedef __attribute__((ext_vector_type(16))) __bf16 v16bf;
typedef __attribute__((ext_vector_type(8)))  __bf16 v8bf;
typedef __attribute__((ext_vector_type(8)))  float  v8f;

#define NNODES 50000
#define NEDGES 800000
#define DD 64
#define HH 128
#define NL 3
#define NG 128

// ---------------- fast tanh: use gfx1250 V_TANH_F32 if the builtin exists ----------
__device__ __forceinline__ float fast_tanh(float x) {
#if __has_builtin(__builtin_amdgcn_tanhf)
  return __builtin_amdgcn_tanhf(x);
#elif __has_builtin(__builtin_amdgcn_tanh_f32)
  return __builtin_amdgcn_tanh_f32(x);
#else
  // branch-free: tanh(x) = 1 - 2/(exp2(2*log2e*x) + 1); correct at +/-inf.
  float t = __builtin_amdgcn_exp2f(x * 2.88539008177792681f);
  return 1.0f - 2.0f * __builtin_amdgcn_rcpf(t + 1.0f);
#endif
}

// ---------------- weight pack: f32 [K x N] row-major -> bf16 WMMA B-fragments ----
// Fragment layout: [kt][nt][lane 0..31][e 0..15] bf16, matching the ISA 16-bit
// operand layout: lane half selects k+8, e0..7 -> k0+e, e8..15 -> k0+16+(e-8).
__global__ void pack_w_kernel(const float* __restrict__ W, __bf16* __restrict__ Wp,
                              int K, int Ncols) {
  int t = blockIdx.x * 256 + threadIdx.x;
  if (t >= K * Ncols) return;
  int ntiles = Ncols >> 4;
  int e    = t & 15;
  int lane = (t >> 4) & 31;
  int f    = t >> 9;                 // 512 values per fragment
  int nt = f % ntiles;
  int kt = f / ntiles;
  int n     = nt * 16 + (lane & 15);
  int khalf = lane >> 4;
  int k = kt * 32 + khalf * 8 + (e & 7) + ((e >> 3) << 4);
  Wp[t] = (__bf16)W[k * Ncols + n];
}

__global__ void zero_kernel(float* __restrict__ p, int n) {
  int t = blockIdx.x * 256 + threadIdx.x;
  if (t < n) p[t] = 0.0f;
}

// ---------------- edge scatter-add: agg[dst] += x[src], 16 lanes x float4 / edge --
__global__ void scatter_kernel(const float* __restrict__ x, const int* __restrict__ src,
                               const int* __restrict__ dst, float* __restrict__ agg) {
  int t = blockIdx.x * 256 + threadIdx.x;
  int e = t >> 4;
  if (e >= NEDGES) return;
  int q = (t & 15) * 4;
  int s = src[e], d = dst[e];
  float4 v = *(const float4*)(x + s * DD + q);
  float* a = agg + d * DD + q;
  atomicAdd(a + 0, v.x); atomicAdd(a + 1, v.y);
  atomicAdd(a + 2, v.z); atomicAdd(a + 3, v.w);
}

// ---------------- fused per-wave 3-stage MLP with WMMA -----------------------------
__device__ __forceinline__ v16bf lds_afrag(const __bf16* Srow, int k0) {
  union { v16bf v; v8bf h[2]; } u;
  u.h[0] = *(const v8bf*)(Srow + k0);        // k0 .. k0+7
  u.h[1] = *(const v8bf*)(Srow + k0 + 16);   // k0+16 .. k0+23
  return u.v;
}

#define SSTRIDE 136  // bf16 elems; 272B row stride -> 16B aligned, bank-staggered

__global__ __launch_bounds__(256)
void mlp_kernel(const float* __restrict__ x_in, const float* __restrict__ agg,
                const __bf16* __restrict__ W1p, const float* __restrict__ b1,
                const __bf16* __restrict__ W2p, const float* __restrict__ b2,
                const __bf16* __restrict__ W3p, const float* __restrict__ b3,
                float* __restrict__ x_out) {
  __shared__ __bf16 Sh[8][16][SSTRIDE];
  const int wave = threadIdx.x >> 5;
  const int lane = threadIdx.x & 31;
  const int tile = blockIdx.x * 8 + wave;
  const bool active = (tile * 16) < NNODES;   // NNODES % 16 == 0, whole-tile guard
  const int rowBase = tile * 16;
  const int m     = lane & 15;
  const int khalf = lane >> 4;
  const int row   = rowBase + m;
  __bf16 (*S)[SSTRIDE] = Sh[wave];

  // ---- stage 1: (x + agg)[16x64] @ W1[64x128], tanh -> S ----
  if (active) {
    v16bf a1[2];
#pragma unroll
    for (int kt = 0; kt < 2; ++kt) {
      int k0 = kt * 32 + khalf * 8;
      const float* xr = x_in + row * DD + k0;
      const float* ar = agg  + row * DD + k0;
      v16bf a;
#pragma unroll
      for (int h = 0; h < 2; ++h) {
        float4 xv0 = *(const float4*)(xr + h * 16);
        float4 xv1 = *(const float4*)(xr + h * 16 + 4);
        float4 av0 = *(const float4*)(ar + h * 16);
        float4 av1 = *(const float4*)(ar + h * 16 + 4);
        a[h * 8 + 0] = (__bf16)(xv0.x + av0.x);
        a[h * 8 + 1] = (__bf16)(xv0.y + av0.y);
        a[h * 8 + 2] = (__bf16)(xv0.z + av0.z);
        a[h * 8 + 3] = (__bf16)(xv0.w + av0.w);
        a[h * 8 + 4] = (__bf16)(xv1.x + av1.x);
        a[h * 8 + 5] = (__bf16)(xv1.y + av1.y);
        a[h * 8 + 6] = (__bf16)(xv1.z + av1.z);
        a[h * 8 + 7] = (__bf16)(xv1.w + av1.w);
      }
      a1[kt] = a;
    }
#pragma unroll
    for (int nt = 0; nt < 8; ++nt) {
      v8f c = {};
#pragma unroll
      for (int kt = 0; kt < 2; ++kt) {
        v16bf b = *(const v16bf*)(W1p + (((kt * 8 + nt) * 32 + lane) << 4));
        c = __builtin_amdgcn_wmma_f32_16x16x32_bf16(false, a1[kt], false, b,
                                                    (short)0, c, false, false);
      }
      int col = nt * 16 + (lane & 15);
      float bb = b1[col];
#pragma unroll
      for (int r = 0; r < 8; ++r)
        S[r + khalf * 8][col] = (__bf16)fast_tanh(c[r] + bb);
    }
  }
  __syncthreads();

  // ---- stage 2: S[16x128] @ W2[128x128], tanh -> S (A fully in regs first) ----
  if (active) {
    v16bf a2[4];
#pragma unroll
    for (int kt = 0; kt < 4; ++kt)
      a2[kt] = lds_afrag(&S[m][0], kt * 32 + khalf * 8);
#pragma unroll
    for (int nt = 0; nt < 8; ++nt) {
      v8f c = {};
#pragma unroll
      for (int kt = 0; kt < 4; ++kt) {
        v16bf b = *(const v16bf*)(W2p + (((kt * 8 + nt) * 32 + lane) << 4));
        c = __builtin_amdgcn_wmma_f32_16x16x32_bf16(false, a2[kt], false, b,
                                                    (short)0, c, false, false);
      }
      int col = nt * 16 + (lane & 15);
      float bb = b2[col];
#pragma unroll
      for (int r = 0; r < 8; ++r)
        S[r + khalf * 8][col] = (__bf16)fast_tanh(c[r] + bb);
    }
  }
  __syncthreads();

  // ---- stage 3: S[16x128] @ W3[128x64], tanh(tanh(.)) -> x_out ----
  if (active) {
    v16bf a3[4];
#pragma unroll
    for (int kt = 0; kt < 4; ++kt)
      a3[kt] = lds_afrag(&S[m][0], kt * 32 + khalf * 8);
#pragma unroll
    for (int nt = 0; nt < 4; ++nt) {
      v8f c = {};
#pragma unroll
      for (int kt = 0; kt < 4; ++kt) {
        v16bf b = *(const v16bf*)(W3p + (((kt * 4 + nt) * 32 + lane) << 4));
        c = __builtin_amdgcn_wmma_f32_16x16x32_bf16(false, a3[kt], false, b,
                                                    (short)0, c, false, false);
      }
      int col = nt * 16 + (lane & 15);
      float bb = b3[col];
#pragma unroll
      for (int r = 0; r < 8; ++r) {
        float v = fast_tanh(fast_tanh(c[r] + bb));
        x_out[(rowBase + r + khalf * 8) * DD + col] = v;
      }
    }
  }
}

// ---------------- graph mean pooling ----------------------------------------------
__global__ void pool_sum_kernel(const float* __restrict__ x, const int* __restrict__ batch,
                                float* __restrict__ sums, float* __restrict__ counts) {
  int t = blockIdx.x * 256 + threadIdx.x;
  int i = t >> 4;
  if (i >= NNODES) return;
  int q = (t & 15) * 4;
  int g = batch[i];
  float4 v = *(const float4*)(x + i * DD + q);
  float* s = sums + g * DD + q;
  atomicAdd(s + 0, v.x); atomicAdd(s + 1, v.y);
  atomicAdd(s + 2, v.z); atomicAdd(s + 3, v.w);
  if (q == 0) atomicAdd(counts + g, 1.0f);
}

__global__ void pool_div_kernel(const float* __restrict__ sums,
                                const float* __restrict__ counts,
                                float* __restrict__ out) {
  int t = blockIdx.x * 256 + threadIdx.x;
  if (t >= NG * DD) return;
  out[t] = sums[t] / fmaxf(counts[t / DD], 1.0f);
}

// ---------------- launch -----------------------------------------------------------
extern "C" void kernel_launch(void* const* d_in, const int* in_sizes, int n_in,
                              void* d_out, int out_size, void* d_ws, size_t ws_size,
                              hipStream_t stream) {
  const float* attrs = (const float*)d_in[0];
  const float* W1    = (const float*)d_in[1];
  const float* b1    = (const float*)d_in[2];
  const float* W2    = (const float*)d_in[3];
  const float* b2    = (const float*)d_in[4];
  const float* W3    = (const float*)d_in[5];
  const float* b3    = (const float*)d_in[6];
  const int*   eidx  = (const int*)d_in[7];
  const int*   batch = (const int*)d_in[8];
  const int* esrc = eidx;
  const int* edst = eidx + NEDGES;

  char* ws = (char*)d_ws;
  float*  agg    = (float*)ws;  ws += (size_t)NNODES * DD * 4;
  float*  xbuf   = (float*)ws;  ws += (size_t)NNODES * DD * 4;
  __bf16* W1p    = (__bf16*)ws; ws += (size_t)NL * DD * HH * 2;
  __bf16* W2p    = (__bf16*)ws; ws += (size_t)NL * HH * HH * 2;
  __bf16* W3p    = (__bf16*)ws; ws += (size_t)NL * HH * DD * 2;
  float*  sums   = (float*)ws;  ws += (size_t)NG * DD * 4;
  float*  counts = (float*)ws;  ws += (size_t)NG * 4;

  // pack all layer weights to bf16 B-fragment layout
  for (int l = 0; l < NL; ++l) {
    pack_w_kernel<<<(DD * HH + 255) / 256, 256, 0, stream>>>(W1 + l * DD * HH, W1p + l * DD * HH, DD, HH);
    pack_w_kernel<<<(HH * HH + 255) / 256, 256, 0, stream>>>(W2 + l * HH * HH, W2p + l * HH * HH, HH, HH);
    pack_w_kernel<<<(HH * DD + 255) / 256, 256, 0, stream>>>(W3 + l * HH * DD, W3p + l * HH * DD, HH, DD);
  }

  const int mlpBlocks = (NNODES / 16 + 7) / 8;   // 16-row tile per wave, 8 waves/block
  for (int l = 0; l < NL; ++l) {
    const float* xin = (l == 0) ? attrs : xbuf;
    zero_kernel<<<(NNODES * DD + 255) / 256, 256, 0, stream>>>(agg, NNODES * DD);
    scatter_kernel<<<(NEDGES * 16 + 255) / 256, 256, 0, stream>>>(xin, esrc, edst, agg);
    mlp_kernel<<<mlpBlocks, 256, 0, stream>>>(xin, agg,
        W1p + l * DD * HH, b1 + l * HH,
        W2p + l * HH * HH, b2 + l * HH,
        W3p + l * HH * DD, b3 + l * DD,
        xbuf);
  }

  zero_kernel<<<(NG * DD + NG + 255) / 256, 256, 0, stream>>>(sums, NG * DD + NG); // sums+counts contiguous
  pool_sum_kernel<<<(NNODES * 16 + 255) / 256, 256, 0, stream>>>(xbuf, batch, sums, counts);
  pool_div_kernel<<<(NG * DD + 255) / 256, 256, 0, stream>>>(sums, counts, (float*)d_out);
}
